// _NonLocal2_75909251989682
// MI455X (gfx1250) — compile-verified, matrix operand order checked by reading, not run on hardware
//
#include <hip/hip_runtime.h>
#include <math.h>

// ---------------- CDNA5 WMMA plumbing (wave32, 16x16x32 bf16 -> f32) --------
typedef __attribute__((ext_vector_type(16))) __bf16 v16bf;
typedef __attribute__((ext_vector_type(8)))  float  v8f;
typedef __attribute__((ext_vector_type(4)))  int    v4i;

#define CWH   262144           // c*w*h == w*h*z == 64*4096
#define WHPOS 4096             // w*h
#define NBF   262144.0f        // BN / pool-mean element count per channel

// ---- async global->LDS (GLOBAL_LOAD_ASYNC_TO_LDS_B128, ASYNCcnt) -----------
#if defined(__has_builtin)
#if __has_builtin(__builtin_amdgcn_global_load_async_to_lds_b128) && \
    __has_builtin(__builtin_amdgcn_s_wait_asynccnt)
#define HAS_ASYNC_LDS 1
#endif
#endif

typedef __attribute__((address_space(1))) v4i* gv4i_p;   // global int4*
typedef __attribute__((address_space(3))) v4i* lv4i_p;   // LDS int4*

__device__ __forceinline__ void cp16_g2l(void* lds, const void* g) {
#ifdef HAS_ASYNC_LDS
  __builtin_amdgcn_global_load_async_to_lds_b128((gv4i_p)(g), (lv4i_p)(lds), 0, 0);
#else
  *(float4*)lds = *(const float4*)g;
#endif
}
__device__ __forceinline__ void cp_wait() {
#ifdef HAS_ASYNC_LDS
  __builtin_amdgcn_s_wait_asynccnt(0);
#endif
}

// K position of packed element j for lane-half hi8 (0 or 8), per ISA 7.12.2
__device__ __forceinline__ int kmap(int j, int hi8) {
  return ((j >> 3) << 4) + hi8 + (j & 7);
}

// A-style gather: row fixed per lane (M or N), K varies with j, row-major [row][pitch]
__device__ __forceinline__ v16bf fragA(const float* s, int pitch, int row, int kbase, int hi8) {
  v16bf f;
  const float* p = s + row * pitch + kbase;
#pragma unroll
  for (int j = 0; j < 16; ++j) f[j] = (__bf16)p[kmap(j, hi8)];
  return f;
}

// B-style gather: col fixed per lane (N), K varies with j, data stored [K][pitch]
__device__ __forceinline__ v16bf fragB(const float* s, int pitch, int col, int kbase, int hi8) {
  v16bf f;
  const float* p = s + col + kbase * pitch;
#pragma unroll
  for (int j = 0; j < 16; ++j) f[j] = (__bf16)p[kmap(j, hi8) * pitch];
  return f;
}

__device__ __forceinline__ v8f wmma_bf16(v16bf a, v16bf b, v8f c) {
  return __builtin_amdgcn_wmma_f32_16x16x32_bf16(false, a, false, b, (short)0, c, false, false);
}

// ---------------- kernel 0: zero small accumulators -------------------------
__global__ void k_init(float* acc) {
  if (threadIdx.x < 384) acc[threadIdx.x] = 0.f;   // psum[128] bnsum[128] bnsq[128]
}

// ---------------- kernel 1: p[z] partial sums (coalesced + atomics) ---------
__global__ __launch_bounds__(256) void k_psum(const float* __restrict__ x, float* __restrict__ psum) {
  __shared__ float s[64];
  const int b = blockIdx.y, tid = threadIdx.x;
  if (tid < 64) s[tid] = 0.f;
  __syncthreads();
  const int z = tid & 63, qoff = tid >> 6;                 // 4 q-rows in flight
  const float* xb = x + (size_t)b * 64 * CWH;
  const size_t qbase = (size_t)blockIdx.x * 1024;
  float acc = 0.f;
  for (int i = 0; i < 256; ++i) acc += xb[(qbase + qoff + 4 * (size_t)i) * 64 + z];
  atomicAdd(&s[z], acc);
  __syncthreads();
  if (tid < 64) atomicAdd(&psum[b * 64 + tid], s[tid]);
}

// ---------------- kernel 2: g/phi projection GEMMs + bias + z-pair maxpool --
// pitches: weight tiles 68, X tile 36 (multiples of 4 => 16B rows for async b128;
// A gathers hit banks (4*row+k)%64, distinct over the 16 rows of a lane group)
__global__ __launch_bounds__(256) void k_proj(const float* __restrict__ x,
                                              const float* __restrict__ psum,
                                              const float* __restrict__ g_w,
                                              const float* __restrict__ g_b,
                                              const float* __restrict__ phi_w,
                                              const float* __restrict__ phi_b,
                                              float* __restrict__ g_pool,
                                              float* __restrict__ phi_pool) {
  __shared__ float GW[64 * 68];
  __shared__ float PW[64 * 68];
  __shared__ float Xs[64 * 36];   // 64 channels x 32 z-columns
  __shared__ float ps[64];
  const int b = blockIdx.y, tid = threadIdx.x;
  const int n0 = blockIdx.x * 32;                 // 32 consecutive z of one (w,h)
  const int pos = n0 >> 6, zbase = n0 & 63;

  // async-stage weights (1024 b128 rows each) and the X tile (512 b128)
#pragma unroll
  for (int it = 0; it < 4; ++it) {
    int i = tid + it * 256, r = i >> 4, seg = (i & 15) * 4;
    cp16_g2l(&GW[r * 68 + seg], &g_w[r * 64 + seg]);
    cp16_g2l(&PW[r * 68 + seg], &phi_w[r * 64 + seg]);
  }
  const float* xb = x + (size_t)b * 64 * CWH;
#pragma unroll
  for (int it = 0; it < 2; ++it) {
    int i = tid + it * 256, c = i >> 3, seg = (i & 7) * 4;
    cp16_g2l(&Xs[c * 36 + seg], &xb[(size_t)c * CWH + n0 + seg]);
  }
  if (tid < 64) ps[tid] = psum[b * 64 + tid] * (1.f / NBF);
  cp_wait();
  __syncthreads();

  const int wave = tid >> 5, lane = tid & 31;
  const int l16 = lane & 15, hi8 = (lane & 16) ? 8 : 0;
  const int mt = wave >> 1, nt = wave & 1;        // 4 M-tiles x 2 N-tiles = 8 waves
  const int col = nt * 16 + l16;                  // z within the 32-wide tile
  const float pz = ps[zbase + col];               // uniform per lane across all K

  v8f accg = {}; v8f accp = {};
#pragma unroll
  for (int kb = 0; kb < 64; kb += 32) {
    v16bf ag = fragA(GW, 68, mt * 16 + l16, kb, hi8);
    v16bf ap = fragA(PW, 68, mt * 16 + l16, kb, hi8);
    v16bf bg, bp;
#pragma unroll
    for (int j = 0; j < 16; ++j) {
      float v = Xs[(kb + kmap(j, hi8)) * 36 + col];
      bg[j] = (__bf16)v;
      bp[j] = (__bf16)(v * pz);
    }
    accg = wmma_bf16(ag, bg, accg);
    accp = wmma_bf16(ap, bp, accp);
  }

  const int z = zbase + col;
#pragma unroll
  for (int r = 0; r < 8; ++r) {
    int M = mt * 16 + r + hi8;
    float vg = accg[r] + g_b[M];
    float vp = accp[r] + phi_b[M];
    vg = fmaxf(vg, __shfl_xor(vg, 1));            // z-pair max across adjacent lanes
    vp = fmaxf(vp, __shfl_xor(vp, 1));
    if ((z & 1) == 0) {
      // g_pool: (b, pos, nz, ic) so attention can async-stage contiguous rows
      g_pool[(((size_t)b * WHPOS + pos) * 32 + (z >> 1)) * 64 + M] = vg;
      // phi_pool: (b, ic, pos, nz)
      phi_pool[(((size_t)b * 64 + M) * WHPOS + pos) * 32 + (z >> 1)] = vp;
    }
  }
}

// ---------------- kernel 3: per-position attention + fused W-proj + BN stats
__global__ __launch_bounds__(256) void k_attn(const float* __restrict__ x,
                                              const float* __restrict__ psum,
                                              const float* __restrict__ g_pool,
                                              const float* __restrict__ phi_pool,
                                              const float* __restrict__ W_w,
                                              const float* __restrict__ W_b,
                                              float* __restrict__ out,
                                              float* __restrict__ bnsum,
                                              float* __restrict__ bnsq) {
  __shared__ float B0[64 * 68];   // theta[z][c], later W_w[c0][i]
  __shared__ float B1[64 * 36];   // phi[c][nz]
  __shared__ float B2[32 * 68];   // g[nz][ic]
  __shared__ float B3[64 * 33];   // f / softmax a  [z][nz]
  __shared__ float B4[64 * 65];   // y[z][ic]
  __shared__ float ps[64];
  __shared__ float bns[64], bnq[64];
  const int b = blockIdx.y, pos = blockIdx.x, tid = threadIdx.x;
  const int wave = tid >> 5, lane = tid & 31;
  const int l16 = lane & 15, hi8 = (lane & 16) ? 8 : 0;

  if (tid < 64) { ps[tid] = psum[b * 64 + tid] * (1.f / NBF); bns[tid] = 0.f; bnq[tid] = 0.f; }

  // async-stage phi (512 b128) and g (512 b128); overlap with theta compute below
#pragma unroll
  for (int it = 0; it < 2; ++it) {
    int i = tid + it * 256;
    { int c = i >> 3, seg = (i & 7) * 4;
      cp16_g2l(&B1[c * 36 + seg],
               &phi_pool[(((size_t)b * 64 + c) * WHPOS + pos) * 32 + seg]); }
    { int nz = i >> 4, seg = (i & 15) * 4;
      cp16_g2l(&B2[nz * 68 + seg],
               &g_pool[(((size_t)b * WHPOS + pos) * 32 + nz) * 64 + seg]); }
  }
  __syncthreads();                                // ps ready for theta

  const float* xb = x + (size_t)b * 64 * CWH;
  for (int i = tid; i < 4096; i += 256) {         // theta[z][c] = x[c,pos,z] * p[z]
    int c = i >> 6, z = i & 63;
    B0[z * 68 + c] = xb[((size_t)c * WHPOS + pos) * 64 + z] * ps[z];
  }
  cp_wait();
  __syncthreads();

  { // f = theta(64x64) @ phi(64x32): 4 mt x 2 nt tiles over 8 waves
    const int mt = wave >> 1, nt = wave & 1;
    v8f acc = {};
#pragma unroll
    for (int kb = 0; kb < 64; kb += 32)
      acc = wmma_bf16(fragA(B0, 68, mt * 16 + l16, kb, hi8),
                      fragB(B1, 36, nt * 16 + l16, kb, hi8), acc);
#pragma unroll
    for (int r = 0; r < 8; ++r)
      B3[(mt * 16 + r + hi8) * 33 + nt * 16 + l16] = acc[r];
  }
  __syncthreads();

  // softmax over nz (t<64); meanwhile async-stage W_w into B0 (theta is dead)
#pragma unroll
  for (int it = 0; it < 4; ++it) {
    int i = tid + it * 256, r = i >> 4, seg = (i & 15) * 4;
    cp16_g2l(&B0[r * 68 + seg], &W_w[r * 64 + seg]);
  }
  if (tid < 64) {
    float* row = &B3[tid * 33];
    float m = row[0];
    for (int j = 1; j < 32; ++j) m = fmaxf(m, row[j]);
    float s = 0.f;
    for (int j = 0; j < 32; ++j) { float e = __expf(row[j] - m); row[j] = e; s += e; }
    float inv = 1.f / s;
    for (int j = 0; j < 32; ++j) row[j] *= inv;
  }
  cp_wait();
  __syncthreads();

  { // y = a(64x32) @ g(32x64): 16 tiles, 8 waves x 2 passes, single K-step
#pragma unroll
    for (int pass = 0; pass < 2; ++pass) {
      const int mt = wave >> 1, nt = (wave & 1) + 2 * pass;
      v8f acc = {};
      acc = wmma_bf16(fragA(B3, 33, mt * 16 + l16, 0, hi8),
                      fragB(B2, 68, nt * 16 + l16, 0, hi8), acc);
#pragma unroll
      for (int r = 0; r < 8; ++r)
        B4[(mt * 16 + r + hi8) * 65 + nt * 16 + l16] = acc[r];
    }
  }
  __syncthreads();

  { // wy = W(64x64) @ y^T(64x64): B[k=i][n=z] = y[z][i] -> A-style gather on B4
#pragma unroll
    for (int pass = 0; pass < 2; ++pass) {
      const int mt = wave >> 1, nt = (wave & 1) + 2 * pass;
      v8f acc = {};
#pragma unroll
      for (int kb = 0; kb < 64; kb += 32)
        acc = wmma_bf16(fragA(B0, 68, mt * 16 + l16, kb, hi8),
                        fragA(B4, 65, nt * 16 + l16, kb, hi8), acc);
      const int z = nt * 16 + l16;
#pragma unroll
      for (int r = 0; r < 8; ++r) {
        int M = mt * 16 + r + hi8;
        float v = acc[r] + W_b[M];
        out[(((size_t)b * 64 + M) * WHPOS + pos) * 64 + z] = v;
        atomicAdd(&bns[M], v);
        atomicAdd(&bnq[M], v * v);
      }
    }
  }
  __syncthreads();
  if (tid < 64) {
    atomicAdd(&bnsum[b * 64 + tid], bns[tid]);
    atomicAdd(&bnsq[b * 64 + tid], bnq[tid]);
  }
}

// ---------------- kernel 4: BN normalize + affine + residual + spacing select
__global__ __launch_bounds__(256) void k_final(const float* __restrict__ x,
                                               const float* __restrict__ spacings,
                                               const float* __restrict__ bnsum,
                                               const float* __restrict__ bnsq,
                                               const float* __restrict__ gamma,
                                               const float* __restrict__ beta,
                                               float* __restrict__ out) {
  size_t i4 = (size_t)blockIdx.x * blockDim.x + threadIdx.x;   // 8388608 float4s
  if (i4 >= 8388608u) return;
  size_t e = i4 * 4;
  int b = (int)(e >> 24);
  int c = (int)((e >> 18) & 63);
  bool use = spacings[b * 3 + 2] <= 1.5f;
  float mu = bnsum[b * 64 + c] * (1.f / NBF);
  float var = bnsq[b * 64 + c] * (1.f / NBF) - mu * mu;
  float rs = rsqrtf(var + 1e-5f) * gamma[c];
  float be = beta[c];
  float4 wy = reinterpret_cast<float4*>(out)[i4];
  float4 xv = reinterpret_cast<const float4*>(x)[i4];
  float4 r;
  r.x = use ? ((wy.x - mu) * rs + be + xv.x) : xv.x;
  r.y = use ? ((wy.y - mu) * rs + be + xv.y) : xv.y;
  r.z = use ? ((wy.z - mu) * rs + be + xv.z) : xv.z;
  r.w = use ? ((wy.w - mu) * rs + be + xv.w) : xv.w;
  reinterpret_cast<float4*>(out)[i4] = r;
}

// ---------------------------------------------------------------------------
extern "C" void kernel_launch(void* const* d_in, const int* in_sizes, int n_in,
                              void* d_out, int out_size, void* d_ws, size_t ws_size,
                              hipStream_t stream) {
  const float* x        = (const float*)d_in[0];
  const float* spacings = (const float*)d_in[1];
  const float* g_w      = (const float*)d_in[2];
  const float* g_b      = (const float*)d_in[3];
  const float* phi_w    = (const float*)d_in[4];
  const float* phi_b    = (const float*)d_in[5];
  const float* W_w      = (const float*)d_in[6];
  const float* W_b      = (const float*)d_in[7];
  const float* gamma    = (const float*)d_in[8];
  const float* beta     = (const float*)d_in[9];
  float* out = (float*)d_out;

  float* acc      = (float*)d_ws;                 // psum[128] | bnsum[128] | bnsq[128]
  float* psum     = acc;
  float* bnsum    = acc + 128;
  float* bnsq     = acc + 256;
  float* g_pool   = (float*)((char*)d_ws + 4096);           // 64 MiB
  float* phi_pool = g_pool + (size_t)2 * 64 * WHPOS * 32;   // 64 MiB

  k_init <<<1, 384, 0, stream>>>(acc);
  k_psum <<<dim3(256, 2),   dim3(256), 0, stream>>>(x, psum);
  k_proj <<<dim3(8192, 2),  dim3(256), 0, stream>>>(x, psum, g_w, g_b, phi_w, phi_b,
                                                    g_pool, phi_pool);
  k_attn <<<dim3(4096, 2),  dim3(256), 0, stream>>>(x, psum, g_pool, phi_pool,
                                                    W_w, W_b, out, bnsum, bnsq);
  k_final<<<dim3(32768),    dim3(256), 0, stream>>>(x, spacings, bnsum, bnsq,
                                                    gamma, beta, out);
}
